// GraphClassifiction_5695126634841
// MI455X (gfx1250) — matrix operand order repaired and checked by reading, not compiled, not run on hardware
//
#include <hip/hip_runtime.h>
#include <hip/hip_bf16.h>
#include <math.h>

typedef __attribute__((ext_vector_type(16))) __bf16 v16bf;
typedef __attribute__((ext_vector_type(8)))  __bf16 v8bf;
typedef __attribute__((ext_vector_type(8)))  float  v8f;

// ---------------------------------------------------------------------------
// Sizes (fixed by the reference)
// ---------------------------------------------------------------------------
#define NB        640          // 128*5 fused batch
#define TT        32           // temporal length
#define CIN_V     2048
#define CIN_T     768
#define NCH_V     ((CIN_V * 3) / 32)    // 192 bf16-WMMA K-chunks
#define NCH_T     ((CIN_T * 3) / 32)    // 72
#define CPK_V     (CIN_V / 32)          // 64 ci-windows
#define CPK_T     (CIN_T / 32)          // 24
#define PACKV_N   (NCH_V * 2 * 32 * 16) // 196608 bf16
#define PACKT_N   (NCH_T * 2 * 32 * 16) // 73728 bf16
#define NEL       (NB * TT)    // 20480 (b,l) head elements
#define NBLK_H    80           // head grid: 80 blocks * 256 = 20480

#define FXW       1152         // bf16 elems per X stage buffer (34 rows x 32, padded)
#define BSW       3072         // bf16 elems per B stage buffer (3 taps x 2 tiles x 512)
#define FWAVE     1184         // f32 elems of per-wave LDS region (>= 2*FXW bf16, >= 1024 f32)

__device__ __forceinline__ v8f wmma_bf16(v16bf a, v16bf b, v8f c) {
  // v_wmma_f32_16x16x32_bf16  (D = A x B + C), f32 accumulate
  return __builtin_amdgcn_wmma_f32_16x16x32_bf16(false, a, false, b,
                                                 (short)0, c, false, false);
}

__device__ __forceinline__ v8f zero8() {
  v8f z;
  #pragma unroll
  for (int i = 0; i < 8; ++i) z[i] = 0.0f;
  return z;
}

// Async global->LDS 16B copy (CDNA5 GLOBAL_LOAD_ASYNC_TO_LDS_B128, ASYNCcnt).
// ldsOff = byte offset within workgroup LDS (low 32 bits of the flat address).
__device__ __forceinline__ void async_g2l_b128(unsigned ldsOff, const void* gsrc) {
  asm volatile("global_load_async_to_lds_b128 %0, %1, off"
               :: "v"(ldsOff), "v"((unsigned long long)(uintptr_t)gsrc)
               : "memory");
}
__device__ __forceinline__ void wait_asynccnt0() {
  asm volatile("s_wait_asynccnt 0x0" ::: "memory");
}

// ---------------------------------------------------------------------------
// Weight prepack: f32 conv weight (Cout=32, Cin, 3) -> bf16 B-fragments.
// Fragment layout (B 32x16 16-bit, contiguous k order used by this code): lane L holds column
// N = L&15, slot j holds K = ((L>=16)?16:0)+j within the 32-wide chunk.
// K global ordering: k = kk*Cin + ci  (kk = conv tap).
// Storage index: ((c*2 + ntile)*32 + L)*16 + j,  chunk c = kk*CPK + cc.
// ---------------------------------------------------------------------------
__global__ void prepack_kernel(const float* __restrict__ wv,
                               const float* __restrict__ wt,
                               __bf16* __restrict__ packV,
                               __bf16* __restrict__ packT) {
  int idx = blockIdx.x * blockDim.x + threadIdx.x;
  if (idx < PACKV_N) {
    int j = idx & 15, L = (idx >> 4) & 31, nt = (idx >> 9) & 1, c = idx >> 10;
    int K  = c * 32 + ((L >= 16) ? 16 : 0) + j;
    int co = nt * 16 + (L & 15);
    int kk = K / CIN_V, ci = K % CIN_V;
    packV[idx] = (__bf16)wv[(co * CIN_V + ci) * 3 + kk];
  } else if (idx < PACKV_N + PACKT_N) {
    int t = idx - PACKV_N;
    int j = t & 15, L = (t >> 4) & 31, nt = (t >> 9) & 1, c = t >> 10;
    int K  = c * 32 + ((L >= 16) ? 16 : 0) + j;
    int co = nt * 16 + (L & 15);
    int kk = K / CIN_T, ci = K % CIN_T;
    packT[t] = (__bf16)wt[(co * CIN_T + ci) * 3 + kk];
  }
}

// --- stage helpers ---------------------------------------------------------
__device__ __forceinline__ void stage_B(const __bf16* __restrict__ pack,
                                        __bf16* __restrict__ BsBuf,
                                        int cc, int tid, int tapStrideB) {
  const char* base = (const char*)pack + (size_t)cc * 2048;
  #pragma unroll
  for (int u = 0; u < 3; ++u) {
    const int tr  = tid + u * 128;      // 0..383 (384 x 16B = 6KB)
    const int kkq = tr >> 7;            // tap 0..2
    const int o   = (tr & 127) * 16;    // byte offset within 2KB chunk
    unsigned dst  = (unsigned)(uintptr_t)((char*)BsBuf + tr * 16);
    async_g2l_b128(dst, base + (size_t)kkq * tapStrideB + o);
  }
}

template <int CIN>
__device__ __forceinline__ void stage_X(const float* __restrict__ Xrow,
                                        __bf16* __restrict__ FxBuf,
                                        int cc, int lane) {
  const float4* p4 = (const float4*)(Xrow + cc * 32);
  float4 q0 = p4[0], q1 = p4[1], q2 = p4[2], q3 = p4[3];
  float4 q4 = p4[4], q5 = p4[5], q6 = p4[6], q7 = p4[7];
  v16bf t0, t1;
  t0[0]  = (__bf16)q0.x; t0[1]  = (__bf16)q0.y; t0[2]  = (__bf16)q0.z; t0[3]  = (__bf16)q0.w;
  t0[4]  = (__bf16)q1.x; t0[5]  = (__bf16)q1.y; t0[6]  = (__bf16)q1.z; t0[7]  = (__bf16)q1.w;
  t0[8]  = (__bf16)q2.x; t0[9]  = (__bf16)q2.y; t0[10] = (__bf16)q2.z; t0[11] = (__bf16)q2.w;
  t0[12] = (__bf16)q3.x; t0[13] = (__bf16)q3.y; t0[14] = (__bf16)q3.z; t0[15] = (__bf16)q3.w;
  t1[0]  = (__bf16)q4.x; t1[1]  = (__bf16)q4.y; t1[2]  = (__bf16)q4.z; t1[3]  = (__bf16)q4.w;
  t1[4]  = (__bf16)q5.x; t1[5]  = (__bf16)q5.y; t1[6]  = (__bf16)q5.z; t1[7]  = (__bf16)q5.w;
  t1[8]  = (__bf16)q6.x; t1[9]  = (__bf16)q6.y; t1[10] = (__bf16)q6.z; t1[11] = (__bf16)q6.w;
  t1[12] = (__bf16)q7.x; t1[13] = (__bf16)q7.y; t1[14] = (__bf16)q7.z; t1[15] = (__bf16)q7.w;
  __bf16* rp = FxBuf + (lane + 1) * 32;   // row l -> buffer row l+1
  *(v16bf*)rp        = t0;
  *(v16bf*)(rp + 16) = t1;
}

// ---------------------------------------------------------------------------
// conv1d(k=3,pad=1) as im2col bf16 WMMA GEMM, ci-window outer / tap inner,
// double-buffered LDS pipeline:
//   iter cc:  s_wait_asynccnt 0 ; barrier ; issue stage(cc+1 -> buf^1)
//             (async B + reg-converted X) ; 12 WMMAs from buf
// Zero rows at l=-1/32 in both X buffers make A-fragment reads branch-free.
// Then bias+ReLU -> f32 LDS, row L2-normalize, 32x32 Gram via one more
// K=32 bf16 WMMA pass.
// A fragment layout (A 16x32 16-bit, same contiguous k order as B): lane L holds row M = L&15,
// slot j holds K = j + (L>=16?16:0)  -> one contiguous 16-span (two 8-elem loads).
// ---------------------------------------------------------------------------
template <int CIN, int CPKc>
__device__ __forceinline__ void conv_gram(const float* __restrict__ X,
                                          const __bf16* __restrict__ pack,
                                          const float* __restrict__ bias,
                                          int batch, int lane, int tid,
                                          float* __restrict__ Fn,   // per-wave FWAVE f32
                                          __bf16* __restrict__ Bs2, // shared 2*BSW bf16
                                          v8f g[4]) {
  const int l15 = lane & 15;
  const int lh  = lane >> 4;
  const int klo = lh ? 16 : 0;
  __bf16* Fx = (__bf16*)Fn;   // two FXW-elem X buffers at 0 and FXW

  // zero-pad rows for l = -1 (row 0) and l = 32 (row 33) in both buffers
  if (lane < 4) {
    v16bf z;
    #pragma unroll
    for (int j = 0; j < 16; ++j) z[j] = (__bf16)0.0f;
    __bf16* zp = Fx + (lane >> 1) * FXW + ((lane & 1) ? 33 : 0) * 32;
    *(v16bf*)zp = z;
    *(v16bf*)(zp + 16) = z;
  }

  v8f acc[2][2];
  #pragma unroll
  for (int mt = 0; mt < 2; ++mt)
    #pragma unroll
    for (int nt = 0; nt < 2; ++nt) acc[mt][nt] = zero8();

  const float* Xrow = X + ((size_t)batch * TT + lane) * CIN;  // lane stages row==lane

  // prologue: stage window 0 into buffer 0
  stage_B(pack, Bs2, 0, tid, CPKc * 2048);
  stage_X<CIN>(Xrow, Fx, 0, lane);

  for (int cc = 0; cc < CPKc; ++cc) {
    const int p = cc & 1;
    wait_asynccnt0();    // this wave's stage(cc) asyncs done
    __syncthreads();     // all waves: stage(cc) visible, compute(cc-1) finished

    if (cc + 1 < CPKc) { // overlap: stage next window while computing this one
      stage_B(pack, Bs2 + (p ^ 1) * BSW, cc + 1, tid, CPKc * 2048);
      stage_X<CIN>(Xrow, Fx + (p ^ 1) * FXW, cc + 1, lane);
    }

    const __bf16* Bsb = Bs2 + p * BSW;
    const __bf16* Fxb = Fx + p * FXW;
    #pragma unroll
    for (int kk = 0; kk < 3; ++kk) {
      v16bf b0 = *(const v16bf*)(Bsb + (kk * 2 + 0) * 512 + lane * 16);
      v16bf b1 = *(const v16bf*)(Bsb + (kk * 2 + 1) * 512 + lane * 16);
      #pragma unroll
      for (int mt = 0; mt < 2; ++mt) {
        const __bf16* rp = Fxb + (mt * 16 + l15 + kk) * 32 + klo;  // row = l+1
        v8bf lo = *(const v8bf*)rp;
        v8bf hi = *(const v8bf*)(rp + 8);
        v16bf a = __builtin_shufflevector(lo, hi, 0, 1, 2, 3, 4, 5, 6, 7,
                                                  8, 9, 10, 11, 12, 13, 14, 15);
        acc[mt][0] = wmma_bf16(a, b0, acc[mt][0]);
        acc[mt][1] = wmma_bf16(a, b1, acc[mt][1]);
      }
    }
  }
  __syncthreads();

  // bias + ReLU -> Fn (C/D layout: d[r] of lane L is [mt*16+lh*8+r][nt*16+l15])
  #pragma unroll
  for (int mt = 0; mt < 2; ++mt)
    #pragma unroll
    for (int nt = 0; nt < 2; ++nt) {
      const int col  = nt * 16 + l15;
      const float bc = bias[col];
      #pragma unroll
      for (int r = 0; r < 8; ++r) {
        const int row = mt * 16 + lh * 8 + r;
        Fn[row * 32 + col] = fmaxf(acc[mt][nt][r] + bc, 0.0f);
      }
    }
  __syncthreads();

  // row-wise L2 normalize (lane == row)
  {
    float s = 0.0f;
    #pragma unroll 8
    for (int k = 0; k < 32; ++k) { float v = Fn[lane * 32 + k]; s += v * v; }
    float inv = 1.0f / fmaxf(sqrtf(s), 1e-12f);
    #pragma unroll 8
    for (int k = 0; k < 32; ++k) Fn[lane * 32 + k] *= inv;
  }
  __syncthreads();

  // Gram = Fn * Fn^T, K = 32 channels, one bf16 WMMA per 16x16 tile
  v16bf ga[2], gb[2];
  #pragma unroll
  for (int mt = 0; mt < 2; ++mt) {
    const int row = mt * 16 + l15;
    #pragma unroll
    for (int j = 0; j < 16; ++j) {
      const int K = lh * 16 + j;
      ga[mt][j] = (__bf16)Fn[row * 32 + K];
    }
  }
  #pragma unroll
  for (int nt = 0; nt < 2; ++nt) {
    const int row = nt * 16 + l15;       // B[k][n] = Fn[n][k]
    #pragma unroll
    for (int j = 0; j < 16; ++j)
      gb[nt][j] = (__bf16)Fn[row * 32 + lh * 16 + j];
  }
  v8f z = zero8();
  g[0] = wmma_bf16(ga[0], gb[0], z);
  g[1] = wmma_bf16(ga[0], gb[1], z);
  g[2] = wmma_bf16(ga[1], gb[0], z);
  g[3] = wmma_bf16(ga[1], gb[1], z);
}

// ---------------------------------------------------------------------------
// Main fused kernel: 1 wave == 1 batch (grid 160 x 128 threads = 640 waves)
// ---------------------------------------------------------------------------
__global__ __launch_bounds__(128) void fused_main_kernel(
    const float* __restrict__ xv, const float* __restrict__ xt,
    const __bf16* __restrict__ packV, const __bf16* __restrict__ packT,
    const float* __restrict__ bv, const float* __restrict__ bt,
    const float* __restrict__ fa,
    float* __restrict__ out_av, float* __restrict__ out_at,
    float* __restrict__ out_fused, float* __restrict__ maxes) {
  __shared__ __align__(64) float  ldsF[4][FWAVE];     // per-wave X stages / Fn
  __shared__ __align__(64) __bf16 ldsB[2 * BSW];      // double-buffered B stage
  __shared__ float ldsR[4][32];
  const int tid   = threadIdx.x;
  const int wave  = tid >> 5;
  const int lane  = tid & 31;
  const int batch = blockIdx.x * 4 + wave;   // exactly 640
  const int l15   = lane & 15, lh = lane >> 4;

  v8f gv[4], gt[4];
  conv_gram<CIN_V, CPK_V>(xv, packV, bv, batch, lane, tid, ldsF[wave], ldsB, gv);
  conv_gram<CIN_T, CPK_T>(xt, packT, bt, batch, lane, tid, ldsF[wave], ldsB, gt);

  float far[16];
  #pragma unroll
  for (int i = 0; i < 16; ++i) far[i] = fa[i];

  float lmax = 0.0f;
  float* oav = out_av    + (size_t)batch * 1024;
  float* oat = out_at    + (size_t)batch * 1024;
  float* ofu = out_fused + (size_t)batch * 1024;

  #pragma unroll
  for (int mt = 0; mt < 2; ++mt)
    #pragma unroll
    for (int nt = 0; nt < 2; ++nt) {
      const int t = mt * 2 + nt;
      #pragma unroll
      for (int r = 0; r < 8; ++r) {
        const int i = mt * 16 + lh * 8 + r;
        const int j = nt * 16 + l15;
        const float av = fmaxf(gv[t][r], 0.0f);
        const float at = fmaxf(gt[t][r], 0.0f);
        oav[i * 32 + j] = av;
        oat[i * 32 + j] = at;
        const float eye = (i == j) ? 1.0f : 0.0f;
        const float pv0 = eye, pv1 = av, pv2 = av * av, pv3 = pv2 * av;
        const float pt0 = eye, pt1 = at, pt2 = at * at, pt3 = pt2 * at;
        float f =
            pv0 * (far[0]  * pt0 + far[1]  * pt1 + far[2]  * pt2 + far[3]  * pt3) +
            pv1 * (far[4]  * pt0 + far[5]  * pt1 + far[6]  * pt2 + far[7]  * pt3) +
            pv2 * (far[8]  * pt0 + far[9]  * pt1 + far[10] * pt2 + far[11] * pt3) +
            pv3 * (far[12] * pt0 + far[13] * pt1 + far[14] * pt2 + far[15] * pt3);
        f = fmaxf(f, 0.0f);
        ofu[i * 32 + j] = f;
        lmax = fmaxf(lmax, f);
      }
    }
  ldsR[wave][lane] = lmax;
  __syncthreads();
  if (lane == 0) {
    float m = 0.0f;
    #pragma unroll
    for (int k = 0; k < 32; ++k) m = fmaxf(m, ldsR[wave][k]);
    maxes[batch] = m;
  }
}

// ---------------------------------------------------------------------------
// Global max -> 1/(max + 0.01)    (fixed-order, deterministic)
// ---------------------------------------------------------------------------
__global__ void max_reduce_kernel(const float* __restrict__ maxes,
                                  float* __restrict__ stats) {
  __shared__ float red[256];
  float m = 0.0f;
  for (int i = threadIdx.x; i < NB; i += 256) m = fmaxf(m, maxes[i]);
  red[threadIdx.x] = m;
  __syncthreads();
  for (int s = 128; s > 0; s >>= 1) {
    if (threadIdx.x < s) red[threadIdx.x] = fmaxf(red[threadIdx.x], red[threadIdx.x + s]);
    __syncthreads();
  }
  if (threadIdx.x == 0) stats[0] = 1.0f / (red[0] + 0.01f);
}

// ---------------------------------------------------------------------------
// Head pass 1: scale fused in place, x1 = W1 * x + b1, BN1 partial sums.
// thread t -> (b = t/32, l = t%32); x[b,c,l] = fused[b,l,c] (contiguous row).
// ---------------------------------------------------------------------------
__global__ void head1_kernel(float* __restrict__ fused,
                             const float* __restrict__ w1,
                             const float* __restrict__ b1,
                             float* __restrict__ x1,
                             float* __restrict__ part1,
                             const float* __restrict__ stats) {
  const int t   = blockIdx.x * 256 + threadIdx.x;   // 0..20479
  const float inv = stats[0];
  float* p = fused + (size_t)t * 32;
  float d0 = 0.0f, d1 = 0.0f;
  #pragma unroll 8
  for (int c = 0; c < 32; ++c) {
    float v = p[c] * inv;
    p[c] = v;                       // a_fused output, scaled in place
    d0 += w1[c] * v;
    d1 += w1[32 + c] * v;
  }
  d0 += b1[0];
  d1 += b1[1];
  x1[t]       = d0;
  x1[NEL + t] = d1;

  __shared__ float red[4][256];
  const int tid = threadIdx.x;
  red[0][tid] = d0; red[1][tid] = d0 * d0;
  red[2][tid] = d1; red[3][tid] = d1 * d1;
  __syncthreads();
  for (int s = 128; s > 0; s >>= 1) {
    if (tid < s) {
      #pragma unroll
      for (int k = 0; k < 4; ++k) red[k][tid] += red[k][tid + s];
    }
    __syncthreads();
  }
  if (tid == 0) {
    #pragma unroll
    for (int k = 0; k < 4; ++k) part1[blockIdx.x * 4 + k] = red[k][0];
  }
}

__global__ void stats1_kernel(const float* __restrict__ part1,
                              const float* __restrict__ g1,
                              const float* __restrict__ be1,
                              float* __restrict__ stats) {
  __shared__ float red[4][128];
  const int tid = threadIdx.x;
  float s[4] = {0, 0, 0, 0};
  for (int i = tid; i < NBLK_H; i += 128) {
    #pragma unroll
    for (int k = 0; k < 4; ++k) s[k] += part1[i * 4 + k];
  }
  #pragma unroll
  for (int k = 0; k < 4; ++k) red[k][tid] = s[k];
  __syncthreads();
  for (int st = 64; st > 0; st >>= 1) {
    if (tid < st) {
      #pragma unroll
      for (int k = 0; k < 4; ++k) red[k][tid] += red[k][tid + st];
    }
    __syncthreads();
  }
  if (tid == 0) {
    const float N = (float)NEL;
    float m0 = red[0][0] / N, v0 = red[1][0] / N - m0 * m0;
    float m1 = red[2][0] / N, v1 = red[3][0] / N - m1 * m1;
    float sc0 = g1[0] * rsqrtf(v0 + 1e-5f);
    float sc1 = g1[1] * rsqrtf(v1 + 1e-5f);
    stats[1] = sc0; stats[2] = be1[0] - m0 * sc0;
    stats[3] = sc1; stats[4] = be1[1] - m1 * sc1;
  }
}

__global__ void head2_kernel(const float* __restrict__ x1,
                             const float* __restrict__ w2,
                             const float* __restrict__ b2,
                             float* __restrict__ x2,
                             float* __restrict__ part2,
                             const float* __restrict__ stats) {
  const int t = blockIdx.x * 256 + threadIdx.x;
  float h0 = fmaxf(x1[t]       * stats[1] + stats[2], 0.0f);
  float h1 = fmaxf(x1[NEL + t] * stats[3] + stats[4], 0.0f);
  float z  = w2[0] * h0 + w2[1] * h1 + b2[0];
  x2[t] = z;

  __shared__ float red[2][256];
  const int tid = threadIdx.x;
  red[0][tid] = z; red[1][tid] = z * z;
  __syncthreads();
  for (int s = 128; s > 0; s >>= 1) {
    if (tid < s) { red[0][tid] += red[0][tid + s]; red[1][tid] += red[1][tid + s]; }
    __syncthreads();
  }
  if (tid == 0) {
    part2[blockIdx.x * 2 + 0] = red[0][0];
    part2[blockIdx.x * 2 + 1] = red[1][0];
  }
}

__global__ void stats2_kernel(const float* __restrict__ part2,
                              const float* __restrict__ g2,
                              const float* __restrict__ be2,
                              float* __restrict__ stats) {
  __shared__ float red[2][128];
  const int tid = threadIdx.x;
  float s0 = 0.0f, s1 = 0.0f;
  for (int i = tid; i < NBLK_H; i += 128) { s0 += part2[i * 2]; s1 += part2[i * 2 + 1]; }
  red[0][tid] = s0; red[1][tid] = s1;
  __syncthreads();
  for (int st = 64; st > 0; st >>= 1) {
    if (tid < st) { red[0][tid] += red[0][tid + st]; red[1][tid] += red[1][tid + st]; }
    __syncthreads();
  }
  if (tid == 0) {
    const float N = (float)NEL;
    float m = red[0][0] / N, v = red[1][0] / N - m * m;
    float sc = g2[0] * rsqrtf(v + 1e-5f);
    stats[5] = sc; stats[6] = be2[0] - m * sc;
  }
}

__global__ void head3_kernel(const float* __restrict__ x2,
                             const float* __restrict__ w3,
                             const float* __restrict__ b3,
                             float* __restrict__ out_scores,
                             const float* __restrict__ stats) {
  const int t = blockIdx.x * 256 + threadIdx.x;   // 0..4095: bo = t/32, l = t%32
  const int bo = t >> 5, l = t & 31;
  const float sc2 = stats[5], sh2 = stats[6];
  float acc = 0.0f;
  #pragma unroll
  for (int c = 0; c < 5; ++c) {
    float z = x2[(bo * 5 + c) * 32 + l];
    float h = fmaxf(z * sc2 + sh2, 0.0f);
    float s = w3[0] * h + b3[0];
    acc += 1.0f / (1.0f + expf(-s));
  }
  out_scores[t] = acc * 0.2f;
}

// ---------------------------------------------------------------------------
extern "C" void kernel_launch(void* const* d_in, const int* in_sizes, int n_in,
                              void* d_out, int out_size, void* d_ws, size_t ws_size,
                              hipStream_t stream) {
  const float* xv  = (const float*)d_in[0];
  const float* xt  = (const float*)d_in[1];
  const float* wv  = (const float*)d_in[2];
  const float* bv  = (const float*)d_in[3];
  const float* wt  = (const float*)d_in[4];
  const float* bt  = (const float*)d_in[5];
  const float* fa  = (const float*)d_in[6];
  const float* w1  = (const float*)d_in[7];
  const float* b1  = (const float*)d_in[8];
  const float* g1  = (const float*)d_in[9];
  const float* be1 = (const float*)d_in[10];
  const float* w2  = (const float*)d_in[11];
  const float* b2  = (const float*)d_in[12];
  const float* g2  = (const float*)d_in[13];
  const float* be2 = (const float*)d_in[14];
  const float* w3  = (const float*)d_in[15];
  const float* b3  = (const float*)d_in[16];

  float* out        = (float*)d_out;            // tuple, concatenated flat
  float* out_scores = out;                      // (128,32)      = 4096
  float* out_av     = out + 4096;               // (640,32,32)   = 655360
  float* out_at     = out_av + 655360;
  float* out_fused  = out_at + 655360;

  // workspace layout (fully rewritten every call)
  __bf16* packV = (__bf16*)d_ws;
  __bf16* packT = packV + PACKV_N;
  float*  maxes = (float*)(packT + PACKT_N);
  float*  x1    = maxes + NB;
  float*  x2    = x1 + 2 * NEL;
  float*  part1 = x2 + NEL;
  float*  part2 = part1 + NBLK_H * 4;
  float*  stats = part2 + NBLK_H * 2;

  const int packTotal = PACKV_N + PACKT_N;
  prepack_kernel<<<(packTotal + 255) / 256, 256, 0, stream>>>(wv, wt, packV, packT);

  fused_main_kernel<<<NB / 4, 128, 0, stream>>>(xv, xt, packV, packT, bv, bt, fa,
                                                out_av, out_at, out_fused, maxes);

  max_reduce_kernel<<<1, 256, 0, stream>>>(maxes, stats);
  head1_kernel<<<NBLK_H, 256, 0, stream>>>(out_fused, w1, b1, x1, part1, stats);
  stats1_kernel<<<1, 128, 0, stream>>>(part1, g1, be1, stats);
  head2_kernel<<<NBLK_H, 256, 0, stream>>>(x1, w2, b2, x2, part2, stats);
  stats2_kernel<<<1, 128, 0, stream>>>(part2, g2, be2, stats);
  head3_kernel<<<4096 / 256, 256, 0, stream>>>(x2, w3, b3, out_scores, stats);
}
